// AttentionNetwork_9466107920543
// MI455X (gfx1250) — matrix operand order repaired and checked by reading, not compiled, not run on hardware
//
#include <hip/hip_runtime.h>
#include <hip/hip_bf16.h>

typedef __attribute__((ext_vector_type(16))) _Float16 v16h;
typedef __attribute__((ext_vector_type(8)))  _Float16 v8h;
typedef __attribute__((ext_vector_type(8)))  float    v8f;

#define NB 2048
#define LL 512
#define DD 17
#define HH 8
#define HID 256
#define SPECIAL 9
#define OUTD 23

#ifndef __has_builtin
#define __has_builtin(x) 0
#endif
#if __has_builtin(__builtin_amdgcn_global_load_async_to_lds_b32)
#define USE_ASYNC_LDS 1
#else
#define USE_ASYNC_LDS 0
#endif

typedef __attribute__((address_space(1))) int as1_int;
typedef __attribute__((address_space(3))) int as3_int;

// ---------------------------------------------------------------------------
// Prep: qk[h,d] = sum_e q[h,e]*Wk[h,e,d];  qb[h] = sum_e q[h,e]*bk[h,e]
// ---------------------------------------------------------------------------
__global__ void prep_qk_kernel(const float* __restrict__ q,
                               const float* __restrict__ Wk,
                               const float* __restrict__ bk,
                               float* __restrict__ qk, float* __restrict__ qb)
{
    int tid = threadIdx.x;
    if (tid < HH * DD) {
        int h = tid / DD, d = tid % DD;
        float s = 0.f;
        for (int e = 0; e < DD; ++e) s += q[h*DD + e] * Wk[(h*DD + e)*DD + d];
        qk[tid] = s;
    }
    if (tid < HH) {
        float s = 0.f;
        for (int e = 0; e < DD; ++e) s += q[tid*DD + e] * bk[tid*DD + e];
        qb[tid] = s;
    }
}

// ---------------------------------------------------------------------------
// Weight conversions to f16 (packed policy+value, K padded)
// ---------------------------------------------------------------------------
__global__ void conv_w1_kernel(const float* __restrict__ pW1, const float* __restrict__ pb1,
                               const float* __restrict__ vW1, const float* __restrict__ vb1,
                               _Float16* __restrict__ W1all, float* __restrict__ b1all)
{
    int i = blockIdx.x * blockDim.x + threadIdx.x;   // 512*32
    if (i < 512 * 32) {
        int n = i >> 5, k = i & 31;
        float v = 0.f;
        if (k < DD + SPECIAL) v = (n < 256) ? pW1[n*26 + k] : vW1[(n-256)*26 + k];
        W1all[i] = (_Float16)v;
    }
    if (i < 512) b1all[i] = (i < 256) ? pb1[i] : vb1[i - 256];
}

__global__ void conv_w2_kernel(const float* __restrict__ pW2, const float* __restrict__ pb2,
                               const float* __restrict__ vW2, const float* __restrict__ vb2,
                               _Float16* __restrict__ W2all, float* __restrict__ b2all)
{
    int i = blockIdx.x * blockDim.x + threadIdx.x;   // 512*256
    if (i < 512 * 256) {
        int n = i >> 8, k = i & 255;
        float v = (n < 256) ? pW2[n*256 + k] : vW2[(n-256)*256 + k];
        W2all[i] = (_Float16)v;
    }
    if (i < 512) b2all[i] = (i < 256) ? pb2[i] : vb2[i - 256];
}

__global__ void conv_w3_kernel(const float* __restrict__ pW3, const float* __restrict__ pb3,
                               const float* __restrict__ vW3,
                               _Float16* __restrict__ W3p, float* __restrict__ b3p,
                               _Float16* __restrict__ vW3h)
{
    int i = blockIdx.x * blockDim.x + threadIdx.x;   // 32*256
    if (i < 32 * 256) {
        int n = i >> 8, k = i & 255;
        W3p[i] = (_Float16)((n < OUTD) ? pW3[n*256 + k] : 0.f);
    }
    if (i < 256) vW3h[i] = (_Float16)vW3[i];
    if (i < 32)  b3p[i]  = (i < OUTD) ? pb3[i] : 0.f;
}

// ---------------------------------------------------------------------------
// Fused attention: one workgroup (256 thr = 8 waves) per batch row.
// seq async-staged to LDS, wave-per-head softmax, vals collapsed via
// weighted seq sum. Emits comb (attn | special | 0-pad) f16, row stride 32.
// ---------------------------------------------------------------------------
__global__ __launch_bounds__(256) void attn_kernel(
    const float* __restrict__ x,  const float* __restrict__ Wv,
    const float* __restrict__ bv, const float* __restrict__ Wo,
    const float* __restrict__ bo, const float* __restrict__ qk,
    const float* __restrict__ qb, _Float16* __restrict__ comb)
{
    __shared__ float sSeq[LL * DD];     // 34816 B
    __shared__ float sScore[HH * LL];   // 16384 B
    __shared__ float sMask[LL];
    __shared__ float sQk[HH * DD];
    __shared__ float sQb[HH];
    __shared__ float sWs[HH * DD];
    __shared__ float sAgg[HH * DD];

    const int b   = blockIdx.x;
    const int tid = threadIdx.x;
    const float* xb = x + (size_t)b * (LL + 1) * DD;

#if USE_ASYNC_LDS
    // ASYNCcnt-tracked direct global->LDS staging (no VGPR round trip).
    // B32 transfers: x rows are only 4-byte aligned (D=17).
    for (int i = tid; i < LL * DD; i += 256) {
        __builtin_amdgcn_global_load_async_to_lds_b32(
            (as1_int*)(unsigned long long)(xb + DD + i),
            (as3_int*)(unsigned long long)(&sSeq[i]),
            0, 0);
    }
    asm volatile("s_wait_asynccnt 0x0" ::: "memory");
#else
    for (int i = tid; i < LL * DD; i += 256) sSeq[i] = xb[DD + i];
#endif
    for (int i = tid; i < HH * DD; i += 256) sQk[i] = qk[i];
    if (tid < HH) sQb[tid] = qb[tid];
    __syncthreads();

    for (int l = tid; l < LL; l += 256) {
        float s = 0.f;
        #pragma unroll
        for (int d = 0; d < DD; ++d) s += fabsf(sSeq[l*DD + d]);
        sMask[l] = (s != 0.f) ? 1.f : 0.f;
    }
    __syncthreads();

    for (int idx = tid; idx < HH * LL; idx += 256) {
        int h = idx >> 9, l = idx & (LL - 1);
        float s = sQb[h];
        #pragma unroll
        for (int d = 0; d < DD; ++d) s += sSeq[l*DD + d] * sQk[h*DD + d];
        sScore[idx] = (sMask[l] != 0.f) ? s : -1.0e9f;
    }
    __syncthreads();

    // wave-per-head masked softmax (wave32) + weighted seq reduction
    {
        int h = tid >> 5, lane = tid & 31;
        float mx = -3.4e38f;
        for (int l = lane; l < LL; l += 32) mx = fmaxf(mx, sScore[h*LL + l]);
        for (int off = 16; off > 0; off >>= 1) mx = fmaxf(mx, __shfl_xor(mx, off, 32));
        float sum = 0.f;
        for (int l = lane; l < LL; l += 32) sum += __expf(sScore[h*LL + l] - mx);
        for (int off = 16; off > 0; off >>= 1) sum += __shfl_xor(sum, off, 32);
        float inv = 1.0f / sum;
        for (int l = lane; l < LL; l += 32)
            sScore[h*LL + l] = __expf(sScore[h*LL + l] - mx) * inv;

        // ws[h,d] = sum_l w[h,l]*seq[l,d]: per-lane partials over strided l,
        // then 5-step butterfly reduce across the wave (all 32 lanes busy).
        float acc[DD];
        #pragma unroll
        for (int d = 0; d < DD; ++d) acc[d] = 0.f;
        for (int l = lane; l < LL; l += 32) {
            float w = sScore[h*LL + l];
            #pragma unroll
            for (int d = 0; d < DD; ++d) acc[d] += w * sSeq[l*DD + d];
        }
        #pragma unroll
        for (int d = 0; d < DD; ++d)
            for (int off = 16; off > 0; off >>= 1)
                acc[d] += __shfl_xor(acc[d], off, 32);
        if (lane == 0) {
            #pragma unroll
            for (int d = 0; d < DD; ++d) sWs[h*DD + d] = acc[d];
        }
    }
    __syncthreads();

    if (tid < HH * DD) {             // agg[h,e] = Wv[h]@ws[h] + bv[h]
        int h = tid / DD, e = tid % DD;
        float acc = bv[h*DD + e];
        #pragma unroll
        for (int d = 0; d < DD; ++d) acc += Wv[(h*DD + e)*DD + d] * sWs[h*DD + d];
        sAgg[tid] = acc;
    }
    __syncthreads();

    if (tid < DD) {                  // attn = agg @ Wo.T + bo
        float acc = bo[tid];
        for (int t = 0; t < HH * DD; ++t) acc += Wo[tid*(HH*DD) + t] * sAgg[t];
        comb[(size_t)b*32 + tid] = (_Float16)acc;
    } else if (tid < DD + SPECIAL) {
        comb[(size_t)b*32 + tid] = (_Float16)xb[tid - DD];   // special
    } else if (tid < 32) {
        comb[(size_t)b*32 + tid] = (_Float16)0.f;            // K pad
    }
}

// ---------------------------------------------------------------------------
// WMMA fragment helpers (16-bit A 16x32 / B 32x16 in contiguous-K order, f32 C-D layout)
// ---------------------------------------------------------------------------
__device__ __forceinline__ v16h load_a_frag(const _Float16* __restrict__ arow,
                                            int k0, int lg)
{
    // lane group 0: K = k0+{0..15}; group 1: K = k0+{16..31}
    v8h a0 = *(const v8h*)(arow + k0 + 16*lg);
    v8h a1 = *(const v8h*)(arow + k0 + 16*lg + 8);
    v16h a;
    #pragma unroll
    for (int j = 0; j < 8; ++j) { a[j] = a0[j]; a[8+j] = a1[j]; }
    return a;
}

// ---------------------------------------------------------------------------
// GEMM + bias + ReLU -> f16.  C[m,n] = A[m, acol+k] * W[n,k]
// 4 waves/block, one 16x16 tile per wave. aColPerNet selects the policy/value
// half of A for the packed N=512 layout (0 for layer 1, 256 for layer 2).
// ---------------------------------------------------------------------------
__global__ __launch_bounds__(128) void gemm_relu_f16(
    const _Float16* __restrict__ A, int lda, int aColPerNet,
    const _Float16* __restrict__ W, int ldw, int K,
    const float* __restrict__ bias,
    _Float16* __restrict__ out, int ldo)
{
    int wave = threadIdx.x >> 5;
    int lane = threadIdx.x & 31;
    int n0 = (blockIdx.x * 4 + wave) * 16;
    int m0 = blockIdx.y * 16;
    int acol = (n0 >> 8) * aColPerNet;
    int lh = lane & 15, lg = lane >> 4;

    v8f acc = {};
    const _Float16* arow = A + (size_t)(m0 + lh) * lda + acol;
    const _Float16* wrow = W + (size_t)(n0 + lh) * ldw;
    for (int k0 = 0; k0 < K; k0 += 32) {
        v16h a = load_a_frag(arow, k0, lg);
        v16h bf = *(const v16h*)(wrow + k0 + 16*lg);  // B[k,n]=W[n,k], 16 contig K
        acc = __builtin_amdgcn_wmma_f32_16x16x32_f16(
            false, a, false, bf, (short)0, acc, false, false);
    }
    int col = n0 + lh;
    float bb = bias[col];
    int rbase = m0 + 8*lg;
    #pragma unroll
    for (int r = 0; r < 8; ++r) {
        float v = acc[r] + bb;
        v = v > 0.f ? v : 0.f;
        out[(size_t)(rbase + r) * ldo + col] = (_Float16)v;
    }
}

// ---------------------------------------------------------------------------
// Logits head: h2p (cols 0..255 of h2) @ pW3.T (padded 32x256) + b -> f32 out
// ---------------------------------------------------------------------------
__global__ __launch_bounds__(64) void logits_wmma_kernel(
    const _Float16* __restrict__ A,      // h2, ld 512, policy half at col 0
    const _Float16* __restrict__ W3p,    // 32x256 f16 (rows >=23 zero)
    const float* __restrict__ b3p,
    float* __restrict__ out)             // [2048 x 23]
{
    int wave = threadIdx.x >> 5;
    int lane = threadIdx.x & 31;
    int n0 = wave * 16;
    int m0 = blockIdx.x * 16;
    int lh = lane & 15, lg = lane >> 4;

    v8f acc = {};
    const _Float16* arow = A + (size_t)(m0 + lh) * 512;
    const _Float16* wrow = W3p + (size_t)(n0 + lh) * 256;
    for (int k0 = 0; k0 < 256; k0 += 32) {
        v16h a = load_a_frag(arow, k0, lg);
        v16h bf = *(const v16h*)(wrow + k0 + 16*lg);
        acc = __builtin_amdgcn_wmma_f32_16x16x32_f16(
            false, a, false, bf, (short)0, acc, false, false);
    }
    int col = n0 + lh;
    if (col < OUTD) {
        float bb = b3p[col];
        int rbase = m0 + 8*lg;
        #pragma unroll
        for (int r = 0; r < 8; ++r)
            out[(size_t)(rbase + r) * OUTD + col] = acc[r] + bb;
    }
}

// ---------------------------------------------------------------------------
// Value head: per-row dot(h2v[row], vW3) + vb3
// ---------------------------------------------------------------------------
__global__ void value_kernel(const _Float16* __restrict__ h2,
                             const _Float16* __restrict__ vW3h,
                             const float* __restrict__ vb3,
                             float* __restrict__ out)
{
    int r = blockIdx.x * blockDim.x + threadIdx.x;
    if (r >= NB) return;
    const _Float16* p = h2 + (size_t)r * 512 + 256;   // value half
    float acc = vb3[0];
    #pragma unroll 8
    for (int k = 0; k < HID; ++k) acc += (float)p[k] * (float)vW3h[k];
    out[r] = acc;
}

// ---------------------------------------------------------------------------
extern "C" void kernel_launch(void* const* d_in, const int* in_sizes, int n_in,
                              void* d_out, int out_size, void* d_ws, size_t ws_size,
                              hipStream_t stream)
{
    const float* x   = (const float*)d_in[0];
    const float* Wk  = (const float*)d_in[1];
    const float* bk  = (const float*)d_in[2];
    const float* Wv  = (const float*)d_in[3];
    const float* bv  = (const float*)d_in[4];
    const float* q   = (const float*)d_in[5];
    const float* Wo  = (const float*)d_in[6];
    const float* bo  = (const float*)d_in[7];
    const float* pW1 = (const float*)d_in[8];
    const float* pb1 = (const float*)d_in[9];
    const float* pW2 = (const float*)d_in[10];
    const float* pb2 = (const float*)d_in[11];
    const float* pW3 = (const float*)d_in[12];
    const float* pb3 = (const float*)d_in[13];
    const float* vW1 = (const float*)d_in[14];
    const float* vb1 = (const float*)d_in[15];
    const float* vW2 = (const float*)d_in[16];
    const float* vb2 = (const float*)d_in[17];
    const float* vW3 = (const float*)d_in[18];
    const float* vb3 = (const float*)d_in[19];

    char* ws = (char*)d_ws;
    constexpr size_t O_QK   = 0;                    // 136 f32
    constexpr size_t O_QB   = 1024;                 // 8 f32
    constexpr size_t O_COMB = 2048;                 // 2048*32 f16 = 131072
    constexpr size_t O_W1   = O_COMB + 131072;      // 512*32  f16 = 32768
    constexpr size_t O_B1   = O_W1 + 32768;         // 512 f32
    constexpr size_t O_W2   = O_B1 + 2048;          // 512*256 f16 = 262144
    constexpr size_t O_B2   = O_W2 + 262144;        // 512 f32
    constexpr size_t O_W3   = O_B2 + 2048;          // 32*256  f16 = 16384
    constexpr size_t O_B3   = O_W3 + 16384;         // 32 f32
    constexpr size_t O_VW3  = O_B3 + 128;           // 256 f16
    constexpr size_t O_H1   = 450560;               // 2048*512 f16 = 2 MB
    constexpr size_t O_H2   = O_H1 + 2097152;       // 2048*512 f16 = 2 MB

    float*    qk    = (float*)(ws + O_QK);
    float*    qb    = (float*)(ws + O_QB);
    _Float16* comb  = (_Float16*)(ws + O_COMB);
    _Float16* W1all = (_Float16*)(ws + O_W1);
    float*    b1all = (float*)(ws + O_B1);
    _Float16* W2all = (_Float16*)(ws + O_W2);
    float*    b2all = (float*)(ws + O_B2);
    _Float16* W3p   = (_Float16*)(ws + O_W3);
    float*    b3p   = (float*)(ws + O_B3);
    _Float16* vW3h  = (_Float16*)(ws + O_VW3);
    _Float16* h1    = (_Float16*)(ws + O_H1);
    _Float16* h2    = (_Float16*)(ws + O_H2);

    float* logits = (float*)d_out;            // 2048*23
    float* value  = logits + (size_t)NB * OUTD;

    prep_qk_kernel<<<1, 256, 0, stream>>>(q, Wk, bk, qk, qb);
    conv_w1_kernel<<<64, 256, 0, stream>>>(pW1, pb1, vW1, vb1, W1all, b1all);
    conv_w2_kernel<<<512, 256, 0, stream>>>(pW2, pb2, vW2, vb2, W2all, b2all);
    conv_w3_kernel<<<32, 256, 0, stream>>>(pW3, pb3, vW3, W3p, b3p, vW3h);

    attn_kernel<<<NB, 256, 0, stream>>>(x, Wv, bv, Wo, bo, qk, qb, comb);

    // layer1: comb(2048x32) @ W1all(512x32)^T -> h1(2048x512)
    gemm_relu_f16<<<dim3(8, 128), 128, 0, stream>>>(comb, 32, 0,
                                                    W1all, 32, 32, b1all, h1, 512);
    // layer2: h1 (per-net 256 cols) @ W2all(512x256)^T -> h2(2048x512)
    gemm_relu_f16<<<dim3(8, 128), 128, 0, stream>>>(h1, 512, 256,
                                                    W2all, 256, 256, b2all, h2, 512);
    // heads
    logits_wmma_kernel<<<128, 64, 0, stream>>>(h2, W3p, b3p, logits);
    value_kernel<<<8, 256, 0, stream>>>(h2, vW3h, vb3, value);
}
